// MultiScaleAttention_62886911148195
// MI455X (gfx1250) — compile-verified, matrix operand order checked by reading, not run on hardware
//
#include <hip/hip_runtime.h>
#include <hip/hip_bf16.h>

// ---------------------------------------------------------------------------
// MultiScaleAttention on MI455X (gfx1250, wave32, WMMA).
// All GEMM-shaped work (QKV projection, Q.K^T, P.V, fusion) runs on
// v_wmma_f32_16x16x32_bf16 (bf16 in, f32 accumulate).
// ---------------------------------------------------------------------------

#define C_DIM   256
#define HEADS   8
#define HDIM    32
#define HW      64
#define NPIX    4096   // 64*64

typedef __attribute__((ext_vector_type(16))) __bf16 bf16x16;
typedef __attribute__((ext_vector_type(8)))  float  f32x8;

__device__ __forceinline__ f32x8 wmma_bf16(bf16x16 a, bf16x16 b, f32x8 c) {
  // D = A(16x32) * B(32x16) + C(16x16), f32 accumulate
  return __builtin_amdgcn_wmma_f32_16x16x32_bf16(
      /*neg_a=*/false, a, /*neg_b=*/false, b,
      /*c_mod=*/(short)0, c, /*reuse_a=*/false, /*reuse_b=*/false);
}

// Load 8 contiguous bf16 into vector elements [base, base+8).
// ISA 16-bit A/B fragment layout: lane-half h holds K = {8h..8h+7, 8h+16..8h+23}
// => per lane, two contiguous 8-element runs.
__device__ __forceinline__ void load8(bf16x16& v, int base, const __bf16* __restrict__ p) {
#pragma unroll
  for (int i = 0; i < 8; ++i) v[base + i] = p[i];
}

__device__ __forceinline__ int iclampi(int x, int lo, int hi) {
  return x < lo ? lo : (x > hi ? hi : x);
}

// ---------------------------------------------------------------------------
// fp32 -> bf16 convert (weights)
// ---------------------------------------------------------------------------
__global__ void msa_cvt_bf16(const float* __restrict__ in, __bf16* __restrict__ out, int n) {
  int i = blockIdx.x * blockDim.x + threadIdx.x;
  if (i < n) out[i] = (__bf16)in[i];
}

// ---------------------------------------------------------------------------
// Average pool (NCHW), scale s: 64x64 -> Hs x Hs
// ---------------------------------------------------------------------------
__global__ void msa_pool(const float* __restrict__ x, float* __restrict__ xs,
                         int s, int Hs) {
  int idx = blockIdx.x * blockDim.x + threadIdx.x;
  int total = C_DIM * Hs * Hs;
  if (idx >= total) return;
  int c  = idx / (Hs * Hs);
  int rm = idx % (Hs * Hs);
  int y  = rm / Hs, xx = rm % Hs;
  float acc = 0.f;
  for (int dy = 0; dy < s; ++dy)
    for (int dx = 0; dx < s; ++dx)
      acc += x[c * NPIX + (y * s + dy) * HW + (xx * s + dx)];
  xs[idx] = acc / (float)(s * s);
}

// ---------------------------------------------------------------------------
// Fused depthwise 3x3 (dw) + positional depthwise 3x3 (pos), zero-pad SAME.
//  xdwT : bf16, pixel-major [n][c]  (WMMA-B-friendly)
//  posT : f32,  pixel-major [n][c]
// ---------------------------------------------------------------------------
__global__ void msa_dwpos(const float* __restrict__ xs,
                          const float* __restrict__ dww, const float* __restrict__ dwb,
                          const float* __restrict__ posw, const float* __restrict__ posb,
                          __bf16* __restrict__ xdwT, float* __restrict__ posT, int Hs) {
  int idx = blockIdx.x * blockDim.x + threadIdx.x;
  int total = C_DIM * Hs * Hs;
  if (idx >= total) return;
  int c  = idx / (Hs * Hs);
  int rm = idx % (Hs * Hs);
  int y  = rm / Hs, xx = rm % Hs;
  float a0 = dwb[c], a1 = posb[c];
  const float* w0 = dww  + c * 9;
  const float* w1 = posw + c * 9;
#pragma unroll
  for (int ky = -1; ky <= 1; ++ky) {
#pragma unroll
    for (int kx = -1; kx <= 1; ++kx) {
      int yy = y + ky, xc = xx + kx;
      if (yy < 0 || yy >= Hs || xc < 0 || xc >= Hs) continue;  // zero pad
      float v = xs[c * Hs * Hs + yy * Hs + xc];
      int wi = (ky + 1) * 3 + (kx + 1);
      a0 += v * w0[wi];
      a1 += v * w1[wi];
    }
  }
  int n = y * Hs + xx;
  xdwT[n * C_DIM + c] = (__bf16)a0;
  posT[n * C_DIM + c] = a1;
}

// ---------------------------------------------------------------------------
// QKV projection GEMM:  qkv[m, n] = pw_w[m, :] . xdw[:, n] + pw_b[m]
//   M = 768 (q|k|v rows), K = 256, N = Ns.  One wave per 16x16 tile.
// Epilogue: q,k get +pos and go to pixel-major [n][c] bf16 (attention A/B),
//           v goes channel-major [c][n] bf16 (attention PV B-fragment).
// ---------------------------------------------------------------------------
__global__ void msa_qkv_gemm(const __bf16* __restrict__ pwW, const float* __restrict__ pwB,
                             const __bf16* __restrict__ xdwT, const float* __restrict__ posT,
                             __bf16* __restrict__ qT, __bf16* __restrict__ kT,
                             __bf16* __restrict__ vB, int Ns) {
  const int lane = threadIdx.x;
  const int hl = lane >> 4, r = lane & 15;
  const int n0 = blockIdx.x * 16;
  const int m0 = blockIdx.y * 16;

  f32x8 acc = {};
  const __bf16* arow = pwW  + (size_t)(m0 + r) * C_DIM + hl * 8;   // weight row m0+r
  const __bf16* brow = xdwT + (size_t)(n0 + r) * C_DIM + hl * 8;   // pixel row n0+r
#pragma unroll
  for (int kt = 0; kt < 8; ++kt) {                                 // K = 8 * 32
    bf16x16 a, b;
    load8(a, 0, arow); load8(a, 8, arow + 16);
    load8(b, 0, brow); load8(b, 8, brow + 16);
    __builtin_prefetch(arow + 64, 0, 3);                           // global_prefetch_b8
    __builtin_prefetch(brow + 64, 0, 3);
    arow += 32; brow += 32;
    acc = wmma_bf16(a, b, acc);
  }

  const int ncol = n0 + r;  // D: lane <-> N column; VGPR reg <-> M row (reg + 8*half)
#pragma unroll
  for (int reg = 0; reg < 8; ++reg) {
    int m = m0 + reg + 8 * hl;
    float val = acc[reg] + pwB[m];
    if (m < 256) {                                                 // Q (+pos)
      qT[(size_t)ncol * C_DIM + m] = (__bf16)(val + posT[(size_t)ncol * C_DIM + m]);
    } else if (m < 512) {                                          // K (+pos)
      int c = m - 256;
      kT[(size_t)ncol * C_DIM + c] = (__bf16)(val + posT[(size_t)ncol * C_DIM + c]);
    } else {                                                       // V (channel-major)
      int c = m - 512;
      vB[(size_t)c * Ns + ncol] = (__bf16)val;
    }
  }
}

// ---------------------------------------------------------------------------
// Flash attention, one wave per (head, 16-row n-tile).
//   S = (Q^T K) / sqrt(32): single K=32 WMMA per 16x16 tile (head_dim == 32).
//   Online softmax over m; P re-fragmented through LDS; O += P.V via 2 WMMAs.
// ---------------------------------------------------------------------------
__global__ void msa_attn(const __bf16* __restrict__ qT, const __bf16* __restrict__ kT,
                         const __bf16* __restrict__ vB, __bf16* __restrict__ outT, int Ns) {
  __shared__ __bf16 Pl[16 * 32];   // P tile, [row n][col m], for A-fragment reload

  const int lane = threadIdx.x;
  const int hl = lane >> 4, r = lane & 15;
  const int n0 = blockIdx.x * 16;
  const int h  = blockIdx.y;
  const float scale = 0.17677669529663687f;  // 1/sqrt(32)

  // A(Q^T): rows n, K = d (0..31). Loaded once.
  bf16x16 aq;
  {
    const __bf16* p = qT + (size_t)(n0 + r) * C_DIM + h * HDIM + hl * 8;
    load8(aq, 0, p); load8(aq, 8, p + 16);
  }

  f32x8 o0 = {}, o1 = {};
  float rmax[8], rsum[8];
#pragma unroll
  for (int i = 0; i < 8; ++i) { rmax[i] = -1e30f; rsum[i] = 0.f; }

  for (int m0 = 0; m0 < Ns; m0 += 32) {
    // B(K-matrix): K = d, N = m column
    bf16x16 bk0, bk1;
    const __bf16* k0p = kT + (size_t)(m0 + r) * C_DIM + h * HDIM + hl * 8;
    const __bf16* k1p = k0p + (size_t)16 * C_DIM;
    load8(bk0, 0, k0p); load8(bk0, 8, k0p + 16);
    load8(bk1, 0, k1p); load8(bk1, 8, k1p + 16);

    f32x8 z = {};
    f32x8 s0 = wmma_bf16(aq, bk0, z);   // cols m0 + (lane&15)
    f32x8 s1 = wmma_bf16(aq, bk1, z);   // cols m0+16 + (lane&15)

    float alpha[8];
#pragma unroll
    for (int reg = 0; reg < 8; ++reg) {
      float v0 = s0[reg] * scale, v1 = s1[reg] * scale;
      // tile row-max across the 16-lane half (xor masks stay in-half)
      float t = fmaxf(v0, v1);
      t = fmaxf(t, __shfl_xor(t, 1, 32));
      t = fmaxf(t, __shfl_xor(t, 2, 32));
      t = fmaxf(t, __shfl_xor(t, 4, 32));
      t = fmaxf(t, __shfl_xor(t, 8, 32));
      float nm = fmaxf(rmax[reg], t);
      float p0 = __expf(v0 - nm), p1 = __expf(v1 - nm);
      float ts = p0 + p1;
      ts += __shfl_xor(ts, 1, 32);
      ts += __shfl_xor(ts, 2, 32);
      ts += __shfl_xor(ts, 4, 32);
      ts += __shfl_xor(ts, 8, 32);
      float al = __expf(rmax[reg] - nm);
      rsum[reg] = rsum[reg] * al + ts;
      rmax[reg] = nm;
      alpha[reg] = al;
      s0[reg] = p0; s1[reg] = p1;
    }

    __syncthreads();  // protect Pl vs. previous iteration's A-fragment reads
#pragma unroll
    for (int reg = 0; reg < 8; ++reg) {
      Pl[(reg + 8 * hl) * 32 + r]      = (__bf16)s0[reg];
      Pl[(reg + 8 * hl) * 32 + r + 16] = (__bf16)s1[reg];
    }
    __syncthreads();

    // Re-fragment P as A (rows n, K = m 0..31)
    bf16x16 ap;
    {
      const __bf16* pp = &Pl[r * 32 + hl * 8];
      load8(ap, 0, pp); load8(ap, 8, pp + 16);
    }
    // B(V): K = m, N = d.  v stored channel-major => contiguous m per lane.
    bf16x16 bv0, bv1;
    const __bf16* v0p = vB + (size_t)(h * HDIM + r) * Ns + m0 + hl * 8;
    const __bf16* v1p = vB + (size_t)(h * HDIM + 16 + r) * Ns + m0 + hl * 8;
    load8(bv0, 0, v0p); load8(bv0, 8, v0p + 16);
    load8(bv1, 0, v1p); load8(bv1, 8, v1p + 16);

#pragma unroll
    for (int reg = 0; reg < 8; ++reg) { o0[reg] *= alpha[reg]; o1[reg] *= alpha[reg]; }
    o0 = wmma_bf16(ap, bv0, o0);
    o1 = wmma_bf16(ap, bv1, o1);
  }

  // Normalize and store pixel-major [n][c]
#pragma unroll
  for (int reg = 0; reg < 8; ++reg) {
    float inv = 1.0f / rsum[reg];
    int n = n0 + reg + 8 * hl;
    outT[(size_t)n * C_DIM + h * HDIM + r]      = (__bf16)(o0[reg] * inv);
    outT[(size_t)n * C_DIM + h * HDIM + 16 + r] = (__bf16)(o1[reg] * inv);
  }
}

// ---------------------------------------------------------------------------
// Bilinear upsample (half-pixel centers, matching jax.image.resize) + concat
// into pixel-major bf16 cat[p][3*256].
// ---------------------------------------------------------------------------
__global__ void msa_upcat(const __bf16* __restrict__ a0, const __bf16* __restrict__ a1,
                          const __bf16* __restrict__ a2, __bf16* __restrict__ catT) {
  int idx = blockIdx.x * blockDim.x + threadIdx.x;
  const int total = 3 * C_DIM * NPIX;
  if (idx >= total) return;
  int i  = idx / (C_DIM * NPIX);
  int rm = idx % (C_DIM * NPIX);
  int c  = rm / NPIX;
  int p  = rm % NPIX;
  int y = p / HW, xx = p % HW;
  float val;
  if (i == 0) {
    val = (float)a0[(size_t)p * C_DIM + c];
  } else {
    const __bf16* a = (i == 1) ? a1 : a2;
    int s  = (i == 1) ? 2 : 4;
    int Hs = HW / s;
    float fy = (y + 0.5f) / (float)s - 0.5f;
    float fx = (xx + 0.5f) / (float)s - 0.5f;
    int y0 = (int)floorf(fy); float wy = fy - (float)y0;
    int x0 = (int)floorf(fx); float wx = fx - (float)x0;
    int y0c = iclampi(y0, 0, Hs - 1), y1c = iclampi(y0 + 1, 0, Hs - 1);
    int x0c = iclampi(x0, 0, Hs - 1), x1c = iclampi(x0 + 1, 0, Hs - 1);
    float g00 = (float)a[((size_t)(y0c * Hs + x0c)) * C_DIM + c];
    float g01 = (float)a[((size_t)(y0c * Hs + x1c)) * C_DIM + c];
    float g10 = (float)a[((size_t)(y1c * Hs + x0c)) * C_DIM + c];
    float g11 = (float)a[((size_t)(y1c * Hs + x1c)) * C_DIM + c];
    val = (1.f - wy) * ((1.f - wx) * g00 + wx * g01) +
          wy * ((1.f - wx) * g10 + wx * g11);
  }
  catT[(size_t)p * (3 * C_DIM) + i * C_DIM + c] = (__bf16)val;
}

// ---------------------------------------------------------------------------
// Fusion GEMM: out[o, p] = fusion_w[o, :] . cat[:, p] + fusion_b[o]
//   M = 256, K = 768, N = 4096, fp32 output.
// ---------------------------------------------------------------------------
__global__ void msa_fusion_gemm(const __bf16* __restrict__ fW, const float* __restrict__ fB,
                                const __bf16* __restrict__ catT, float* __restrict__ out) {
  const int lane = threadIdx.x;
  const int hl = lane >> 4, r = lane & 15;
  const int n0 = blockIdx.x * 16;   // pixel tile
  const int m0 = blockIdx.y * 16;   // out-channel tile

  f32x8 acc = {};
  const __bf16* arow = fW   + (size_t)(m0 + r) * 768 + hl * 8;
  const __bf16* brow = catT + (size_t)(n0 + r) * 768 + hl * 8;
#pragma unroll
  for (int kt = 0; kt < 24; ++kt) {  // K = 24 * 32
    bf16x16 a, b;
    load8(a, 0, arow); load8(a, 8, arow + 16);
    load8(b, 0, brow); load8(b, 8, brow + 16);
    __builtin_prefetch(brow + 64, 0, 3);
    arow += 32; brow += 32;
    acc = wmma_bf16(a, b, acc);
  }
  const int ncol = n0 + r;
#pragma unroll
  for (int reg = 0; reg < 8; ++reg) {
    int m = m0 + reg + 8 * hl;
    out[(size_t)m * NPIX + ncol] = acc[reg] + fB[m];
  }
}

// ---------------------------------------------------------------------------
// Host launcher
// ---------------------------------------------------------------------------
extern "C" void kernel_launch(void* const* d_in, const int* in_sizes, int n_in,
                              void* d_out, int out_size, void* d_ws, size_t ws_size,
                              hipStream_t stream) {
  const float* x     = (const float*)d_in[0];
  const float* dw_w  = (const float*)d_in[1];
  const float* dw_b  = (const float*)d_in[2];
  const float* pw_w  = (const float*)d_in[3];
  const float* pw_b  = (const float*)d_in[4];
  const float* pos_w = (const float*)d_in[5];
  const float* pos_b = (const float*)d_in[6];
  const float* fu_w  = (const float*)d_in[7];
  const float* fu_b  = (const float*)d_in[8];
  float* out = (float*)d_out;

  char* ws = (char*)d_ws;
  size_t off = 0;
  auto carve = [&](size_t bytes) -> void* {
    void* p = (void*)(ws + off);
    off += (bytes + 255) & ~(size_t)255;
    return p;
  };

  const int NsArr[3] = {4096, 1024, 256};
  const int HsArr[3] = {64, 32, 16};

  float*  xs2  = (float*)carve((size_t)C_DIM * 1024 * sizeof(float));
  float*  xs4  = (float*)carve((size_t)C_DIM * 256 * sizeof(float));
  __bf16* pwWb = (__bf16*)carve((size_t)3 * 768 * 256 * sizeof(__bf16));
  __bf16* fuWb = (__bf16*)carve((size_t)256 * 768 * sizeof(__bf16));

  __bf16* xdwT[3]; float* posT[3];
  __bf16 *qT[3], *kT[3], *vB[3], *aoT[3];
  for (int i = 0; i < 3; ++i) {
    size_t Ns = (size_t)NsArr[i];
    xdwT[i] = (__bf16*)carve(C_DIM * Ns * sizeof(__bf16));
    posT[i] = (float*)carve(C_DIM * Ns * sizeof(float));
    qT[i]   = (__bf16*)carve(C_DIM * Ns * sizeof(__bf16));
    kT[i]   = (__bf16*)carve(C_DIM * Ns * sizeof(__bf16));
    vB[i]   = (__bf16*)carve(C_DIM * Ns * sizeof(__bf16));
    aoT[i]  = (__bf16*)carve(C_DIM * Ns * sizeof(__bf16));
  }
  __bf16* catT = (__bf16*)carve((size_t)NPIX * 3 * C_DIM * sizeof(__bf16));

  // Weight conversion to bf16
  { int n = 3 * 768 * 256; msa_cvt_bf16<<<(n + 255) / 256, 256, 0, stream>>>(pw_w, pwWb, n); }
  { int n = 256 * 768;     msa_cvt_bf16<<<(n + 255) / 256, 256, 0, stream>>>(fu_w, fuWb, n); }

  // Pools
  { int n = C_DIM * 32 * 32; msa_pool<<<(n + 255) / 256, 256, 0, stream>>>(x, xs2, 2, 32); }
  { int n = C_DIM * 16 * 16; msa_pool<<<(n + 255) / 256, 256, 0, stream>>>(x, xs4, 4, 16); }

  const float* xsArr[3] = {x, xs2, xs4};
  for (int i = 0; i < 3; ++i) {
    int Ns = NsArr[i], Hs = HsArr[i];
    int tot = C_DIM * Ns;
    msa_dwpos<<<(tot + 255) / 256, 256, 0, stream>>>(
        xsArr[i], dw_w + (size_t)i * C_DIM * 9, dw_b + (size_t)i * C_DIM,
        pos_w + (size_t)i * C_DIM * 9, pos_b + (size_t)i * C_DIM,
        xdwT[i], posT[i], Hs);

    dim3 gq(Ns / 16, 48);  // 768/16 = 48 M-tiles
    msa_qkv_gemm<<<gq, 32, 0, stream>>>(
        pwWb + (size_t)i * 768 * 256, pw_b + (size_t)i * 768,
        xdwT[i], posT[i], qT[i], kT[i], vB[i], Ns);

    dim3 ga(Ns / 16, HEADS);
    msa_attn<<<ga, 32, 0, stream>>>(qT[i], kT[i], vB[i], aoT[i], Ns);
  }

  { int n = 3 * C_DIM * NPIX;
    msa_upcat<<<(n + 255) / 256, 256, 0, stream>>>(aoT[0], aoT[1], aoT[2], catT); }

  dim3 gf(NPIX / 16, C_DIM / 16);
  msa_fusion_gemm<<<gf, 32, 0, stream>>>(fuWb, fu_b, catT, out);
}
